// SelfAttentionHead_39651138077321
// MI455X (gfx1250) — matrix operand order repaired and checked by reading, not compiled, not run on hardware
//
#include <hip/hip_runtime.h>
#include <stdint.h>

typedef __attribute__((ext_vector_type(16))) _Float16 v16h;
typedef __attribute__((ext_vector_type(8)))  _Float16 v8h;
typedef __attribute__((ext_vector_type(8)))  float    v8f;
typedef __attribute__((ext_vector_type(4)))  float    v4f;
typedef __attribute__((ext_vector_type(4)))  int      v4i;

#define TT 128            // sequence length
#define CC 384            // embedding dim
#define HH 64             // head size
#define NCHUNK (CC / 32)  // 12 K-chunks for the projections

// ---------------- async global->LDS path (CDNA5), with sync fallback -------
#if __has_builtin(__builtin_amdgcn_global_load_async_to_lds_b128) && \
    __has_builtin(__builtin_amdgcn_s_wait_asynccnt)
#define HAVE_ASYNC_LDS 1
#define WAIT_ASYNC(N) __builtin_amdgcn_s_wait_asynccnt(N)
#else
#define HAVE_ASYNC_LDS 0
#define WAIT_ASYNC(N)
#endif

typedef __attribute__((address_space(1))) v4i gas_v4i;   // global int4
typedef __attribute__((address_space(3))) v4i las_v4i;   // LDS int4

// ---- LDS byte offsets (total 147456 B dynamic shared) ----
// ks   : [128][64] f16   @      0  (16 KB)
// vsT  : [64][128] f16   @  16384  (16 KB)   v transposed: [h][t]
// qs   : [128][64] f16   @  32768  (16 KB)
// scor : [128][128] f32  @  49152  (64 KB)
// wei  : [128][128] f16  @ 114688  (32 KB)
// phase-1 double-buffered f32 x chunks alias the score region:
// xch0 : [128][32] f32   @  49152  (16 KB)
// xch1 : [128][32] f32   @  65536  (16 KB)
#define OFF_KS   0
#define OFF_VST  16384
#define OFF_QS   32768
#define OFF_SCOR 49152
#define OFF_XCH0 49152
#define OFF_XCH1 65536
#define OFF_WEI  114688
#define SMEM_BYTES 147456

// Each thread copies 64 contiguous bytes (16 f32) of an x chunk into LDS.
__device__ __forceinline__ void stage_x_chunk(const float* __restrict__ g, float* l) {
#if HAVE_ASYNC_LDS
    gas_v4i* gp = (gas_v4i*)const_cast<float*>(g);
    las_v4i* lp = (las_v4i*)l;
    __builtin_amdgcn_global_load_async_to_lds_b128(gp + 0, lp + 0, 0, 0);
    __builtin_amdgcn_global_load_async_to_lds_b128(gp + 1, lp + 1, 0, 0);
    __builtin_amdgcn_global_load_async_to_lds_b128(gp + 2, lp + 2, 0, 0);
    __builtin_amdgcn_global_load_async_to_lds_b128(gp + 3, lp + 3, 0, 0);
#else
    v4f d0 = *(const v4f*)(g + 0);
    v4f d1 = *(const v4f*)(g + 4);
    v4f d2 = *(const v4f*)(g + 8);
    v4f d3 = *(const v4f*)(g + 12);
    *(v4f*)(l + 0)  = d0;
    *(v4f*)(l + 4)  = d1;
    *(v4f*)(l + 8)  = d2;
    *(v4f*)(l + 12) = d3;
#endif
}

// A-fragment (16x32 f16), gathered from an f32 row in LDS:
//   lane<16 : m=lane,     halfs 0..15 -> K=0..15
//   lane>=16: m=lane-16,  halfs 0..15 -> K=16..31
__device__ __forceinline__ v16h ld_frag_a32(const float* p, int kbA) {
    v4f a0 = *(const v4f*)(p + kbA);
    v4f a1 = *(const v4f*)(p + kbA + 4);
    v4f a2 = *(const v4f*)(p + kbA + 8);
    v4f a3 = *(const v4f*)(p + kbA + 12);
    v16h a;
#pragma unroll
    for (int i = 0; i < 4; ++i) {
        a[i]      = (_Float16)a0[i];
        a[i + 4]  = (_Float16)a1[i];
        a[i + 8]  = (_Float16)a2[i];
        a[i + 12] = (_Float16)a3[i];
    }
    return a;
}

// Same A-fragment gather from an f16 row (used for qs / wei).
__device__ __forceinline__ v16h ld_frag_a(const _Float16* p, int kbA) {
    v8h lo = *(const v8h*)(p + kbA);
    v8h hi = *(const v8h*)(p + kbA + 8);
    v16h a;
#pragma unroll
    for (int i = 0; i < 8; ++i) { a[i] = lo[i]; a[i + 8] = hi[i]; }
    return a;
}

// B-fragment (32x16 f16): this lane's 16 contiguous K values starting at p.
__device__ __forceinline__ v16h ld_frag_b(const _Float16* p) {
    v8h lo = *(const v8h*)(p);
    v8h hi = *(const v8h*)(p + 8);
    v16h b;
#pragma unroll
    for (int i = 0; i < 8; ++i) { b[i] = lo[i]; b[i + 8] = hi[i]; }
    return b;
}

#define WMMA_F16(A, B, Cacc) \
    __builtin_amdgcn_wmma_f32_16x16x32_f16(false, (A), false, (B), (short)0, (Cacc), false, false)

// ---------------------------------------------------------------------------
// Pre-kernel: pack Wq/Wk/Wv (f32, [384][64]) into f16 B-fragment order:
//   wpk[((mat*12 + kk)*4 + nt)*512 + lane*16 + i]
//     = W[(kk*32 + (lane<16?0:16) + i) * 64 + nt*16 + (lane&15)]
// 73728 halfs = 144 KB, lives in d_ws, L2-resident for all 1024 blocks.
// ---------------------------------------------------------------------------
__global__ __launch_bounds__(256)
void pack_w_kernel(const float* __restrict__ Wq, const float* __restrict__ Wk,
                   const float* __restrict__ Wv, _Float16* __restrict__ wpk) {
    int idx  = blockIdx.x * 256 + threadIdx.x;   // 0 .. 73727
    int i    = idx & 15;
    int lane = (idx >> 4) & 31;
    int nt   = (idx >> 9) & 3;
    int rest = idx >> 11;                        // mat*12 + kk
    int kk   = rest % NCHUNK;
    int mat  = rest / NCHUNK;
    int kbB  = (lane < 16) ? 0 : 16;
    int n    = nt * 16 + (lane & 15);
    int k    = kk * 32 + kbB + i;
    const float* W = (mat == 0) ? Wq : (mat == 1) ? Wk : Wv;
    wpk[idx] = (_Float16)W[k * HH + n];
}

// ---------------------------------------------------------------------------
// Main kernel: one workgroup (8 waves) per batch element.
// ---------------------------------------------------------------------------
__global__ __launch_bounds__(256)
void attn_head_kernel(const float* __restrict__ x,
                      const _Float16* __restrict__ wpk,
                      float* __restrict__ out) {
    extern __shared__ char smem[];
    _Float16* ks   = (_Float16*)(smem + OFF_KS);
    _Float16* vsT  = (_Float16*)(smem + OFF_VST);
    _Float16* qs   = (_Float16*)(smem + OFF_QS);
    float*    scor = (float*)   (smem + OFF_SCOR);
    _Float16* wei  = (_Float16*)(smem + OFF_WEI);
    float*    xch0 = (float*)   (smem + OFF_XCH0);
    float*    xch1 = (float*)   (smem + OFF_XCH1);

    const int tid  = threadIdx.x;
    const int wave = tid >> 5;          // 8 waves
    const int lane = tid & 31;
    const int row0 = wave * 16;         // this wave's 16 query rows
    const int mloc = lane & 15;
    const bool lo16 = (lane < 16);
    const int kbA  = lo16 ? 0 : 16;     // A-fragment K base
    const int ncol = lane & 15;         // B/C column within 16-tile
    const int mrowoff = lo16 ? 0 : 8;   // C-fragment row offset

    const float* xb = x + (size_t)blockIdx.x * TT * CC;

    // staging slice for this thread: 16 consecutive floats of the chunk
    const int srow = tid >> 1;
    const int scol = (tid & 1) * 16;
    const float* gstage = xb + srow * CC + scol;
    float* lstage0 = xch0 + srow * 32 + scol;
    float* lstage1 = xch1 + srow * 32 + scol;

    // ---------------- Phase 1: q,k,v = x @ W (K-chunked, double-buffered) --
    v8f qa[4], ka[4], va[4];
    const v8f vzero = {};
#pragma unroll
    for (int nt = 0; nt < 4; ++nt) { qa[nt] = vzero; ka[nt] = vzero; va[nt] = vzero; }

    stage_x_chunk(gstage, lstage0);                      // prefetch chunk 0

    for (int kk = 0; kk < NCHUNK; ++kk) {
        if (kk + 1 < NCHUNK)                             // prefetch chunk kk+1
            stage_x_chunk(gstage + (kk + 1) * 32, (kk & 1) ? lstage0 : lstage1);
        if (kk + 1 < NCHUNK) { WAIT_ASYNC(4); } else { WAIT_ASYNC(0); }
        __syncthreads();                                 // chunk kk visible to all

        const float* xc = (kk & 1) ? xch1 : xch0;
        v16h a = ld_frag_a32(xc + (row0 + mloc) * 32, kbA);

        // B fragments straight from the packed, L2-hot weight table.
        const _Float16* wf0 = wpk + (size_t)kk * 2048 + lane * 16;
#pragma unroll
        for (int mat = 0; mat < 3; ++mat) {
            const _Float16* wf = wf0 + mat * (NCHUNK * 4 * 512);
            v16h b0 = ld_frag_b(wf + 0 * 512);
            v16h b1 = ld_frag_b(wf + 1 * 512);
            v16h b2 = ld_frag_b(wf + 2 * 512);
            v16h b3 = ld_frag_b(wf + 3 * 512);
            v8f* acc = (mat == 0) ? qa : (mat == 1) ? ka : va;
            acc[0] = WMMA_F16(a, b0, acc[0]);
            acc[1] = WMMA_F16(a, b1, acc[1]);
            acc[2] = WMMA_F16(a, b2, acc[2]);
            acc[3] = WMMA_F16(a, b3, acc[3]);
        }
        __syncthreads();                                 // done reading buf kk
    }

    // ---------------- Phase 2: spill q,k (row-major), v (transposed) -------
#pragma unroll
    for (int nt = 0; nt < 4; ++nt) {
#pragma unroll
        for (int r = 0; r < 8; ++r) {
            int mm = row0 + r + mrowoff;
            int hh = nt * 16 + ncol;
            qs[mm * HH + hh]  = (_Float16)qa[nt][r];
            ks[mm * HH + hh]  = (_Float16)ka[nt][r];
            vsT[hh * TT + mm] = (_Float16)va[nt][r];
        }
    }
    __syncthreads();

    // ---------------- Phase 3: scores = q @ k^T ---------------------------
    {
        const int kbB = lo16 ? 0 : 16;
        v8f sc[8];
#pragma unroll
        for (int nt = 0; nt < 8; ++nt) sc[nt] = vzero;
#pragma unroll
        for (int kk = 0; kk < 2; ++kk) {                 // K over H=64
            v16h a = ld_frag_a(qs + (row0 + mloc) * HH + kk * 32, kbA);
#pragma unroll
            for (int nt = 0; nt < 8; ++nt) {             // N over T=128
                v16h b = ld_frag_b(ks + (nt * 16 + ncol) * HH + kk * 32 + kbB);
                sc[nt] = WMMA_F16(a, b, sc[nt]);
            }
        }
#pragma unroll
        for (int nt = 0; nt < 8; ++nt)
#pragma unroll
            for (int r = 0; r < 8; ++r)
                scor[(row0 + r + mrowoff) * TT + nt * 16 + ncol] = sc[nt][r];
    }
    __syncthreads();

    // ---------------- Phase 4: causal softmax (one row per thread) --------
    if (tid < TT) {
        const int t = tid;
        const float scale = 0.125f;                      // 1/sqrt(64)
        float* srow_ = scor + t * TT;
        float mx = -3.0e38f;
        for (int j = 0; j <= t; ++j) mx = fmaxf(mx, srow_[j] * scale);
        float sum = 0.f;
        for (int j = 0; j <= t; ++j) {
            float e = __expf(srow_[j] * scale - mx);
            srow_[j] = e;
            sum += e;
        }
        const float inv = 1.0f / sum;
        _Float16* wrow = wei + t * TT;
        for (int j = 0; j < TT; ++j)
            wrow[j] = (_Float16)((j <= t) ? srow_[j] * inv : 0.f);
    }
    __syncthreads();

    // ---------------- Phase 5: out = wei @ v ------------------------------
    {
        const int kbB = lo16 ? 0 : 16;
        v8f oc[4];
#pragma unroll
        for (int nt = 0; nt < 4; ++nt) oc[nt] = vzero;
#pragma unroll
        for (int kk = 0; kk < 4; ++kk) {                 // K over T=128
            v16h a = ld_frag_a(wei + (row0 + mloc) * TT + kk * 32, kbA);
#pragma unroll
            for (int nt = 0; nt < 4; ++nt) {             // N over H=64
                v16h b = ld_frag_b(vsT + (nt * 16 + ncol) * TT + kk * 32 + kbB);
                oc[nt] = WMMA_F16(a, b, oc[nt]);
            }
        }
        float* ob = out + (size_t)blockIdx.x * TT * HH;
#pragma unroll
        for (int nt = 0; nt < 4; ++nt)
#pragma unroll
            for (int r = 0; r < 8; ++r)
                ob[(row0 + r + mrowoff) * HH + nt * 16 + ncol] = oc[nt][r];
    }
}

extern "C" void kernel_launch(void* const* d_in, const int* in_sizes, int n_in,
                              void* d_out, int out_size, void* d_ws, size_t ws_size,
                              hipStream_t stream) {
    (void)n_in; (void)ws_size; (void)out_size;
    const float* x  = (const float*)d_in[0];
    const float* Wk = (const float*)d_in[1];
    const float* Wq = (const float*)d_in[2];
    const float* Wv = (const float*)d_in[3];
    float* out = (float*)d_out;
    _Float16* wpk = (_Float16*)d_ws;                 // 147456 B packed weights

    const int B = in_sizes[0] / (TT * CC);           // 1024 batches

    // Allow >64KB dynamic LDS (gfx1250 WGP has 320KB). Host attribute set;
    // not a stream op, safe under graph capture; deterministic.
    (void)hipFuncSetAttribute((const void*)attn_head_kernel,
                              hipFuncAttributeMaxDynamicSharedMemorySize, SMEM_BYTES);

    pack_w_kernel<<<(3 * NCHUNK * 4 * 32 * 16) / 256, 256, 0, stream>>>(Wq, Wk, Wv, wpk);
    attn_head_kernel<<<B, 256, SMEM_BYTES, stream>>>(x, wpk, out);
}
